// CrossAttention_13632226198101
// MI455X (gfx1250) — matrix operand order repaired and checked by reading, not compiled, not run on hardware
//
#include <hip/hip_runtime.h>
#include <hip/hip_bf16.h>
#include <math.h>

// Problem constants (from reference): B=4, S=T=2048, D=1024, H=16, HD=64
#define NB  4
#define NS  2048
#define NT  2048
#define ND  1024
#define NH  16
#define NHD 64

typedef __attribute__((ext_vector_type(16))) __bf16 v16bf;
typedef __attribute__((ext_vector_type(8)))  __bf16 v8bf;
typedef __attribute__((ext_vector_type(8)))  float  v8f;
typedef __attribute__((ext_vector_type(4)))  int    v4i;

#define AS1 __attribute__((address_space(1)))
#define AS3 __attribute__((address_space(3)))

#if __has_builtin(__builtin_amdgcn_global_load_async_to_lds_b128)
#define HAVE_ASYNC_LDS 1
#endif

// Copy 16B global -> LDS. Async path uses the CDNA5 ASYNCcnt-tracked DMA
// (GLOBAL_LOAD_ASYNC_TO_LDS_B128); fallback bounces through VGPRs.
static __device__ __forceinline__ void copy16_g2l(const __bf16* g, __bf16* l) {
#ifdef HAVE_ASYNC_LDS
  __builtin_amdgcn_global_load_async_to_lds_b128(
      (AS1 v4i*)g, (AS3 v4i*)l, /*imm offset*/0, /*cpol*/0);
#else
  *(v8bf*)l = *(const v8bf*)g;
#endif
}

static __device__ __forceinline__ void wait_async_lds() {
#ifdef HAVE_ASYNC_LDS
#if __has_builtin(__builtin_amdgcn_s_wait_asynccnt)
  __builtin_amdgcn_s_wait_asynccnt(0);
#else
  __asm__ volatile("s_wait_asynccnt 0" ::: "memory");
#endif
#endif
}

// A-fragment (16x32 bf16): lane m = lane&15, khalf = lane>>4.
// elements 0..7  = K (khalf*16 + 0..7)       -> 16B contiguous
// elements 8..15 = K (khalf*16 + 8..15)  -> 16B contiguous
static __device__ __forceinline__ v16bf make_a_frag(const __bf16* p0, const __bf16* p1) {
  v8bf lo = *(const v8bf*)p0;
  v8bf hi = *(const v8bf*)p1;
  return __builtin_shufflevector(lo, hi, 0,1,2,3,4,5,6,7,8,9,10,11,12,13,14,15);
}

// ---------------------------------------------------------------------------
// fp32 -> bf16 conversion (vectorized x4; all our sizes are multiples of 4)
// ---------------------------------------------------------------------------
__global__ __launch_bounds__(256) void f32_to_bf16_kernel(
    const float* __restrict__ in, __bf16* __restrict__ out, int n) {
  int i = (blockIdx.x * 256 + threadIdx.x) * 4;
  if (i < n) {
    float4 v = *(const float4*)(in + i);
    out[i+0] = (__bf16)v.x;
    out[i+1] = (__bf16)v.y;
    out[i+2] = (__bf16)v.z;
    out[i+3] = (__bf16)v.w;
  }
}

// ---------------------------------------------------------------------------
// Y = X @ W^T  with bf16 WMMA, fp32 accumulate. Compile-time dims:
//   X: (NB*NS) x ND row-major bf16, W: ND x ND row-major bf16.
// Block: 256 threads = 8 waves; block tile 256(M) x 128(N); each wave 32x128.
// W tile (128 x 32k, 8KB) staged in LDS by async-to-LDS DMA, double buffered.
// B-fragment ds loads are software-pipelined one tile ahead of the WMMAs.
// MODE 0: store fp32 row-major to (float*)Y
// MODE 1: store bf16 to (b,h,s,hd) layout   (for Q and K)
// MODE 2: store bf16 to (b,h,hd,s) layout   (transposed V)
// ---------------------------------------------------------------------------
template <int MODE>
__global__ __launch_bounds__(256) void gemm_xwt_kernel(
    const __bf16* __restrict__ X, const __bf16* __restrict__ W,
    void* __restrict__ Y)
{
  // [buf][n-row 0..127][k 0..31] bf16; row stride 64B -> 16B-aligned frags
  __shared__ __attribute__((aligned(128))) __bf16 Bt[2][128 * 32];

  const int tid  = threadIdx.x;
  const int lane = tid & 31;
  const int wave = tid >> 5;
  const int mrow = lane & 15;
  const int hi   = lane >> 4;
  const int m0 = blockIdx.y * 256 + wave * 32;
  const int n0 = blockIdx.x * 128;

  v8f acc[2][8];
#pragma unroll
  for (int rr = 0; rr < 2; ++rr)
#pragma unroll
    for (int j = 0; j < 8; ++j) acc[rr][j] = {};

  const __bf16* xrow0 = X + (size_t)(m0 + mrow) * ND;        // rows m0..m0+15
  const __bf16* xrow1 = X + (size_t)(m0 + 16 + mrow) * ND;   // rows m0+16..m0+31

  // Stage W[n0..n0+127][k0..k0+31] into Bt[buf]: 512 16B chunks, 2 per thread.
  auto stage = [&](int buf, int k0) {
#pragma unroll
    for (int c = 0; c < 2; ++c) {
      int chunk = c * 256 + tid;        // 0..511
      int row   = chunk >> 2;           // 0..127
      int part  = chunk & 3;            // 16B piece within the 64B k-row
      const __bf16* g = W + (size_t)(n0 + row) * ND + k0 + part * 8;
      copy16_g2l(g, &Bt[buf][row * 32 + part * 8]);
    }
  };

  stage(0, 0);
  for (int k0 = 0; k0 < ND; k0 += 32) {
    const int cur = (k0 >> 5) & 1;
    wait_async_lds();      // this wave's staged chunks have landed in LDS
    __syncthreads();       // everyone's chunks landed; prior reads retired
    if (k0 + 32 < ND) stage(cur ^ 1, k0 + 32);

    v16bf a0 = make_a_frag(xrow0 + k0 + hi*16, xrow0 + k0 + hi*16 + 8);
    v16bf a1 = make_a_frag(xrow1 + k0 + hi*16, xrow1 + k0 + hi*16 + 8);

    // Software-pipeline the 8 B-fragment ds loads one tile ahead.
    const __bf16* bbase = &Bt[cur][mrow * 32 + hi * 16];
    v16bf bcur = *(const v16bf*)bbase;
#pragma unroll
    for (int j = 0; j < 8; ++j) {
      v16bf bnext;
      if (j < 7) bnext = *(const v16bf*)(bbase + (j + 1) * (16 * 32));
      acc[0][j] = __builtin_amdgcn_wmma_f32_16x16x32_bf16(
          false, a0, false, bcur, (short)0, acc[0][j], false, false);
      acc[1][j] = __builtin_amdgcn_wmma_f32_16x16x32_bf16(
          false, a1, false, bcur, (short)0, acc[1][j], false, false);
      if (j < 7) bcur = bnext;
    }
  }

  // D layout: lane column n = lane&15; element r -> row m = r + 8*(lane>>4)
#pragma unroll
  for (int rr = 0; rr < 2; ++rr) {
#pragma unroll
    for (int j = 0; j < 8; ++j) {
#pragma unroll
      for (int r = 0; r < 8; ++r) {
        int mm = m0 + rr*16 + r + 8*hi;
        int nn = n0 + j*16 + mrow;
        float v = acc[rr][j][r];
        if (MODE == 0) {
          ((float*)Y)[(size_t)mm * ND + nn] = v;
        } else {
          int b  = mm >> 11, s  = mm & (NS - 1);       // NS = 2048
          int h  = nn >> 6,  hd = nn & (NHD - 1);      // NHD = 64
          size_t idx = (MODE == 1)
              ? ((size_t)(b*NH + h) * NS + s) * NHD + hd
              : ((size_t)(b*NH + h) * NHD + hd) * (size_t)NS + s;
          ((__bf16*)Y)[idx] = (__bf16)v;
        }
      }
    }
  }
}

// ---------------------------------------------------------------------------
// Flash attention. Grid: (T/128, B*H). Block: 256 threads = 8 waves.
// Each wave owns 16 target rows of one (b,h): Q tile 16x64 in 2 A-fragments,
// loops over S in 32-col chunks, online softmax, O accumulated as 4 f32 tiles.
// Q layout (b,h,t,hd); K layout (b,h,s,hd); V layout (b,h,hd,s) [transposed].
// V fragments are loaded BEFORE the softmax so their latency hides under it.
// ---------------------------------------------------------------------------
__global__ __launch_bounds__(256) void flash_attn_kernel(
    const __bf16* __restrict__ Q, const __bf16* __restrict__ Kh,
    const __bf16* __restrict__ Vt, const unsigned char* __restrict__ msk,
    __bf16* __restrict__ O)
{
  // per-wave P staging: 16 rows x 32 cols bf16, stride 40 (80B, 16B aligned)
  __shared__ __attribute__((aligned(32))) __bf16 lds[8][16 * 40];
  const int lane = threadIdx.x & 31;
  const int wave = threadIdx.x >> 5;
  const int mrow = lane & 15;
  const int hi   = lane >> 4;
  const int bh = blockIdx.y;
  const int b  = bh >> 4;      // NH = 16
  const int h  = bh & 15;
  const int t0 = blockIdx.x * 128 + wave * 16;

  const __bf16* qrow = Q + ((size_t)bh * NT + t0 + mrow) * NHD;
  v16bf qa0 = make_a_frag(qrow + hi*16,      qrow + 8 + hi*16);   // hd 0..31
  v16bf qa1 = make_a_frag(qrow + 32 + hi*16, qrow + 40 + hi*16);  // hd 32..63

  float mrun[8], lrun[8];
  v8f o[4];
#pragma unroll
  for (int r = 0; r < 8; ++r) { mrun[r] = -__builtin_inff(); lrun[r] = 0.f; }
#pragma unroll
  for (int j = 0; j < 4; ++j) o[j] = {};

  __bf16* lp = &lds[wave][0];
  const float scale = 0.125f;   // 1/sqrt(HD)

  for (int s0 = 0; s0 < NS; s0 += 32) {
    // ---- scores: 16 x 32 in two WMMA tiles -------------------------------
    v8f sc[2];
#pragma unroll
    for (int c = 0; c < 2; ++c) {
      int scol = s0 + c*16 + mrow;                 // this lane's source column
      const __bf16* kr = Kh + ((size_t)bh * NS + scol) * NHD;
      v16bf kb0 = *(const v16bf*)(kr + hi*16);        // hd 0..31
      v16bf kb1 = *(const v16bf*)(kr + 32 + hi*16);   // hd 32..63
      v8f z = {};
      z = __builtin_amdgcn_wmma_f32_16x16x32_bf16(false, qa0, false, kb0, (short)0, z, false, false);
      z = __builtin_amdgcn_wmma_f32_16x16x32_bf16(false, qa1, false, kb1, (short)0, z, false, false);
      bool mk = msk[b * NS + scol] != 0;           // per-column source mask
#pragma unroll
      for (int r = 0; r < 8; ++r)
        sc[c][r] = mk ? z[r] * scale : -__builtin_inff();
    }

    // ---- V B-fragments: independent of P, issue before the softmax -------
    v16bf vf[4];
#pragma unroll
    for (int j = 0; j < 4; ++j) {
      const __bf16* vr = Vt + ((size_t)bh * NHD + j*16 + mrow) * NS + s0 + hi*16;
      vf[j] = *(const v16bf*)vr;
    }

    // ---- online softmax (row stats reduced across the 16 n-lanes) --------
    float alpha[8];
#pragma unroll
    for (int r = 0; r < 8; ++r) {
      float rm = fmaxf(sc[0][r], sc[1][r]);
#pragma unroll
      for (int d = 1; d < 16; d <<= 1)
        rm = fmaxf(rm, __shfl_xor(rm, d, 32));
      float mo = mrun[r];
      float mn = fmaxf(mo, rm);
      bool dead = (mn == -__builtin_inff());       // fully-masked-so-far row
      float al = dead ? 1.f : __expf(mo - mn);
      float p0 = dead ? 0.f : __expf(sc[0][r] - mn);
      float p1 = dead ? 0.f : __expf(sc[1][r] - mn);
      sc[0][r] = p0; sc[1][r] = p1;
      float rs = p0 + p1;
#pragma unroll
      for (int d = 1; d < 16; d <<= 1)
        rs += __shfl_xor(rs, d, 32);
      lrun[r] = al * lrun[r] + rs;
      mrun[r] = mn;
      alpha[r] = al;
    }
#pragma unroll
    for (int j = 0; j < 4; ++j)
#pragma unroll
      for (int r = 0; r < 8; ++r)
        o[j][r] *= alpha[r];

    // ---- transpose P through LDS: D-layout -> A-fragment -----------------
#pragma unroll
    for (int c = 0; c < 2; ++c)
#pragma unroll
      for (int r = 0; r < 8; ++r)
        lp[(r + 8*hi)*40 + c*16 + mrow] = (__bf16)sc[c][r];
    __asm__ volatile("s_wait_dscnt 0" ::: "memory");   // cross-lane store->load
    v16bf pa = make_a_frag(lp + mrow*40 + hi*16, lp + mrow*40 + hi*16 + 8);

    // ---- O += P (16x32) x V (32x64) --------------------------------------
#pragma unroll
    for (int j = 0; j < 4; ++j)
      o[j] = __builtin_amdgcn_wmma_f32_16x16x32_bf16(
          false, pa, false, vf[j], (short)0, o[j], false, false);
  }

  // ---- normalize and store to (b, t, h*64+hd) bf16 row-major -------------
  float inv[8];
#pragma unroll
  for (int r = 0; r < 8; ++r) inv[r] = 1.f / lrun[r];
#pragma unroll
  for (int j = 0; j < 4; ++j)
#pragma unroll
    for (int r = 0; r < 8; ++r) {
      int t   = t0 + r + 8*hi;
      int col = h*NHD + j*16 + mrow;
      O[((size_t)b * NT + t) * ND + col] = (__bf16)(o[j][r] * inv[r]);
    }
}

// ---------------------------------------------------------------------------
// Host launcher. Workspace usage (~110 MB): bf16 copies of tgt/src/weights,
// Q (b,h,t,hd), K (b,h,s,hd), V^T (b,h,hd,s), attention output (b,t,d) bf16.
// Inputs (setup_inputs order): src, tgt, attention_mask, Wq, Wk, Wv, Wo.
// ---------------------------------------------------------------------------
extern "C" void kernel_launch(void* const* d_in, const int* in_sizes, int n_in,
                              void* d_out, int out_size, void* d_ws, size_t ws_size,
                              hipStream_t stream) {
  const float* src = (const float*)d_in[0];
  const float* tgt = (const float*)d_in[1];
  const unsigned char* amask = (const unsigned char*)d_in[2];  // jax bool -> 1B
  const float* Wq = (const float*)d_in[3];
  const float* Wk = (const float*)d_in[4];
  const float* Wv = (const float*)d_in[5];
  const float* Wo = (const float*)d_in[6];

  const size_t tokN = (size_t)NB * NS * ND;   // 8,388,608 elements
  const size_t wN   = (size_t)ND * ND;        // 1,048,576 elements

  char* ws = (char*)d_ws;
  __bf16* tgt_bf = (__bf16*)ws; ws += tokN * 2;
  __bf16* src_bf = (__bf16*)ws; ws += tokN * 2;
  __bf16* Wq_bf  = (__bf16*)ws; ws += wN * 2;
  __bf16* Wk_bf  = (__bf16*)ws; ws += wN * 2;
  __bf16* Wv_bf  = (__bf16*)ws; ws += wN * 2;
  __bf16* Wo_bf  = (__bf16*)ws; ws += wN * 2;
  __bf16* Qb     = (__bf16*)ws; ws += tokN * 2;  // (b,h,t,hd)
  __bf16* Kb     = (__bf16*)ws; ws += tokN * 2;  // (b,h,s,hd)
  __bf16* Vtb    = (__bf16*)ws; ws += tokN * 2;  // (b,h,hd,s)
  __bf16* Attnb  = (__bf16*)ws; ws += tokN * 2;  // (b,t,d)

  // 1) fp32 -> bf16 conversions
  {
    int blk = (int)((tokN / 4 + 255) / 256);
    f32_to_bf16_kernel<<<blk, 256, 0, stream>>>(tgt, tgt_bf, (int)tokN);
    f32_to_bf16_kernel<<<blk, 256, 0, stream>>>(src, src_bf, (int)tokN);
    int blkw = (int)((wN / 4 + 255) / 256);
    f32_to_bf16_kernel<<<blkw, 256, 0, stream>>>(Wq, Wq_bf, (int)wN);
    f32_to_bf16_kernel<<<blkw, 256, 0, stream>>>(Wk, Wk_bf, (int)wN);
    f32_to_bf16_kernel<<<blkw, 256, 0, stream>>>(Wv, Wv_bf, (int)wN);
    f32_to_bf16_kernel<<<blkw, 256, 0, stream>>>(Wo, Wo_bf, (int)wN);
  }

  // 2) Q/K/V projections (WMMA GEMM with layout scatter)
  dim3 gg(ND / 128, (NB * NT) / 256);   // (8, 32)
  gemm_xwt_kernel<1><<<gg, 256, 0, stream>>>(tgt_bf, Wq_bf, Qb);
  gemm_xwt_kernel<1><<<gg, 256, 0, stream>>>(src_bf, Wk_bf, Kb);
  gemm_xwt_kernel<2><<<gg, 256, 0, stream>>>(src_bf, Wv_bf, Vtb);

  // 3) flash attention
  dim3 ga(NT / 128, NB * NH);           // (16, 64)
  flash_attn_kernel<<<ga, 256, 0, stream>>>(Qb, Kb, Vtb, amask, Attnb);

  // 4) output projection -> fp32 d_out
  gemm_xwt_kernel<0><<<gg, 256, 0, stream>>>(Attnb, Wo_bf, d_out);
}
